// CrossAttention_3109556322893
// MI455X (gfx1250) — compile-verified, matrix operand order checked by reading, not run on hardware
//
#include <hip/hip_runtime.h>
#include <hip/hip_bf16.h>

// ---------------------------------------------------------------------------
// Temporal self-attention with relative position bias (B=4096, N=16, D=512,
// heads=8, dhead=64) for gfx1250 (MI455X).  All matmuls run on
// v_wmma_f32_16x16x32_f16 with f32 accumulation; intermediates are f16.
// Since N=16 <= MAX_REL=16, dist(t,s) = s-t+16 (no clipping), so the
// rel-pos terms become small WMMA GEMMs + register/LDS gathers.
// Round 3: fix async builtin pointer types (v4i addrspace(1/3) pointers).
// ---------------------------------------------------------------------------

typedef __attribute__((ext_vector_type(16))) _Float16 v16h;
typedef __attribute__((ext_vector_type(8)))  float    v8f;
typedef __attribute__((ext_vector_type(4)))  int      v4i;

#define N_HEADS   8
#define D_HEAD    64
#define N_TOK     16
#define D_MODEL   512
#define B_TOT     4096
#define M_TOK     (B_TOT * N_TOK)      // 65536 tokens
#define ATT_SCALE 0.125f               // 64^-0.5

// --- async global->LDS copy (CDNA5 ASYNCcnt path) ---------------------------

#if defined(__has_builtin)
#if __has_builtin(__builtin_amdgcn_global_load_async_to_lds_b128)
#define HAVE_ASYNC_LDS 1
#endif
#endif
#ifndef HAVE_ASYNC_LDS
#define HAVE_ASYNC_LDS 0
#endif

typedef __attribute__((address_space(1))) v4i gv4i_t;   // global v4i*
typedef __attribute__((address_space(3))) v4i lv4i_t;   // LDS v4i*

__device__ __forceinline__ void async_copy_b128(const void* g, void* l) {
#if HAVE_ASYNC_LDS
  __builtin_amdgcn_global_load_async_to_lds_b128(
      (gv4i_t*)(unsigned long long)g, (lv4i_t*)(unsigned long long)l,
      /*offset=*/0, /*cpol=*/0);
#else
  *(uint4*)l = *(const uint4*)g;
#endif
}

__device__ __forceinline__ void async_wait0() {
  // S_WAIT_ASYNCCNT is a plain SOPP op on gfx1250; harmless if no async ops.
  asm volatile("s_wait_asynccnt 0" ::: "memory");
}

// --- WMMA helpers -----------------------------------------------------------

__device__ __forceinline__ v8f wmma16(v16h a, v16h b, v8f c) {
  // D = A(16x32 f16) * B(32x16 f16) + C(16x16 f32)
  return __builtin_amdgcn_wmma_f32_16x16x32_f16(
      /*neg_a=*/false, a, /*neg_b=*/false, b,
      /*c_mod=*/(short)0, c, /*reuse_a=*/false, /*reuse_b=*/false);
}

// Gather one A/B fragment from a row-major LDS row.
// ISA 7.12.2 (16-bit 16x32 A): lane m = lane&15, lh = lane>>4.
// elements 0..7  -> K = kbase + 8*lh + (0..7)
// elements 8..15 -> K = kbase + 16 + 8*lh + (0..7)
// (B fragment uses the same pattern with m -> n; B tiles are staged n-major.)
__device__ __forceinline__ v16h ld_frag(const _Float16* rowp, int kbase, int lh) {
  union { v16h v; uint4 q[2]; } u;
  u.q[0] = *(const uint4*)(rowp + kbase + 8 * lh);
  u.q[1] = *(const uint4*)(rowp + kbase + 16 + 8 * lh);
  return u.v;
}

// --- pack kernels (f32 -> f16, weight re-layout; negligible cost) -----------

__global__ __launch_bounds__(256) void pack_wqkv_kernel(
    const float* __restrict__ Wq, const float* __restrict__ Wk,
    const float* __restrict__ Wv, _Float16* __restrict__ Wt) {
  int idx = blockIdx.x * 256 + threadIdx.x;            // over 1536*512
  if (idx >= 1536 * 512) return;
  int n = idx >> 9, k = idx & 511;                     // Wt is N-major (N x K)
  int which = n >> 9, c = n & 511;
  const float* W = (which == 0) ? Wq : (which == 1) ? Wk : Wv;
  Wt[idx] = (_Float16)W[k * 512 + c];
}

__global__ __launch_bounds__(256) void pack_wo_kernel(
    const float* __restrict__ Wo, _Float16* __restrict__ Wt) {
  int idx = blockIdx.x * 256 + threadIdx.x;            // over 512*512
  if (idx >= 512 * 512) return;
  int n = idx >> 9, k = idx & 511;
  Wt[idx] = (_Float16)Wo[k * 512 + n];                 // N-major
}

__global__ __launch_bounds__(256) void pack_emb_kernel(
    const float* __restrict__ ek, const float* __restrict__ ev,
    _Float16* __restrict__ ekh, _Float16* __restrict__ evh) {
  int idx = blockIdx.x * 256 + threadIdx.x;            // over 33*64
  if (idx >= 33 * 64) return;
  ekh[idx] = (_Float16)ek[idx];
  evh[idx] = (_Float16)ev[idx];
}

// --- tiled WMMA GEMM: C(M,N) = A(M,K) * Bt(N,K)^T ---------------------------
// EPI==0: A is f32 (x), outputs scattered as f16 into head-split Q/K/V.
// EPI==1: A is f16 (attn output, async-staged), output f32 + bias.
// Block tile 128x64x32, 8 waves (4 along M x 2 along N), 2x2 WMMA per wave.
// Double-buffered LDS with async global->LDS staging; one barrier per k-step.

template <int EPI>
__global__ __launch_bounds__(256) void gemm_wmma_kernel(
    const void* __restrict__ Asrc, const _Float16* __restrict__ Bt,
    const float* __restrict__ bias,
    _Float16* __restrict__ qb, _Float16* __restrict__ kb,
    _Float16* __restrict__ vb, float* __restrict__ outf,
    int M, int N, int K) {
  __shared__ _Float16 As[2][128 * 40];  // 128 rows x 32 k, stride 40
  __shared__ _Float16 Bs[2][64 * 40];   // 64 n-rows x 32 k

  const int tid  = threadIdx.x;
  const int lane = tid & 31;
  const int wave = tid >> 5;
  const int wm   = wave & 3;          // wave tile along M (32 rows each)
  const int wn   = wave >> 2;         // wave tile along N (32 cols each)
  const int m0   = blockIdx.y * 128;
  const int n0   = blockIdx.x * 64;
  const int mrow = lane & 15;
  const int lh   = lane >> 4;

  v8f acc[2][2] = {};

  // tile stagers -------------------------------------------------------------
  auto stage_tiles = [&](int k0, int buf) {
    // A tile (128 x 32)
    if (EPI == 0) {
      // f32 source: convert in VGPRs, plain DS stores (visible after barrier)
      const float* Af = (const float*)Asrc;
#pragma unroll
      for (int it = 0; it < 2; ++it) {
        int g = tid + it * 256;                    // 512 groups of 8 halves
        int row = g >> 2, c8 = (g & 3) * 8;
        const float* src = Af + (size_t)(m0 + row) * K + k0 + c8;
        float4 f0 = *(const float4*)src;
        float4 f1 = *(const float4*)(src + 4);
        union { uint4 q; _Float16 h[8]; } u;
        u.h[0] = (_Float16)f0.x; u.h[1] = (_Float16)f0.y;
        u.h[2] = (_Float16)f0.z; u.h[3] = (_Float16)f0.w;
        u.h[4] = (_Float16)f1.x; u.h[5] = (_Float16)f1.y;
        u.h[6] = (_Float16)f1.z; u.h[7] = (_Float16)f1.w;
        *(uint4*)&As[buf][row * 40 + c8] = u.q;
        if (k0 + 32 < K) __builtin_prefetch(src + 32, 0, 1);
      }
    } else {
      const _Float16* Ah = (const _Float16*)Asrc;
#pragma unroll
      for (int it = 0; it < 2; ++it) {
        int g = tid + it * 256;
        int row = g >> 2, c8 = (g & 3) * 8;
        async_copy_b128(Ah + (size_t)(m0 + row) * K + k0 + c8,
                        &As[buf][row * 40 + c8]);
      }
    }
    // B tile (64 n x 32 k) from N-major weights
    {
      int row = tid >> 2, c8 = (tid & 3) * 8;
      async_copy_b128(Bt + (size_t)(n0 + row) * K + k0 + c8,
                      &Bs[buf][row * 40 + c8]);
    }
  };

  stage_tiles(0, 0);

  int buf = 0;
  for (int k0 = 0; k0 < K; k0 += 32, buf ^= 1) {
    async_wait0();        // this wave's async tile parts complete
    __syncthreads();      // whole tile visible to all waves
    if (k0 + 32 < K) stage_tiles(k0 + 32, buf ^ 1);   // overlap with compute

    v16h a0 = ld_frag(&As[buf][(wm * 32 +      mrow) * 40], 0, lh);
    v16h a1 = ld_frag(&As[buf][(wm * 32 + 16 + mrow) * 40], 0, lh);
    v16h b0 = ld_frag(&Bs[buf][(wn * 32 +      mrow) * 40], 0, lh);
    v16h b1 = ld_frag(&Bs[buf][(wn * 32 + 16 + mrow) * 40], 0, lh);
    acc[0][0] = wmma16(a0, b0, acc[0][0]);
    acc[0][1] = wmma16(a0, b1, acc[0][1]);
    acc[1][0] = wmma16(a1, b0, acc[1][0]);
    acc[1][1] = wmma16(a1, b1, acc[1][1]);
  }

  // ---- epilogue ----
#pragma unroll
  for (int mt = 0; mt < 2; ++mt)
#pragma unroll
    for (int nt = 0; nt < 2; ++nt) {
      int col = n0 + wn * 32 + nt * 16 + mrow;
#pragma unroll
      for (int r = 0; r < 8; ++r) {
        int row = m0 + wm * 32 + mt * 16 + r + 8 * lh;
        float v = acc[mt][nt][r];
        if (EPI == 0) {
          // scatter into head-split layout: [(b*8+h)*16 + ntok][dhead]
          int which = col >> 9;
          int c = col & 511;
          int hi = c >> 6, di = c & 63;
          int b = row >> 4, ntok = row & 15;
          _Float16* dst = (which == 0) ? qb : (which == 1) ? kb : vb;
          dst[((size_t)((b * 8 + hi) * 16 + ntok) << 6) + di] = (_Float16)v;
        } else {
          outf[(size_t)row * N + col] = v + bias[col];
        }
      }
    }
}

// --- attention core: one wave per (b,h); 14 WMMAs per wave ------------------

__global__ __launch_bounds__(128) void attention_kernel(
    const _Float16* __restrict__ qg, const _Float16* __restrict__ kg,
    const _Float16* __restrict__ vg, const _Float16* __restrict__ ekh,
    const _Float16* __restrict__ evh, _Float16* __restrict__ outh) {
  __shared__ _Float16 ek_s[32 * 64];    // emb_k rows 0..31 (dist-major)
  __shared__ _Float16 evt_s[64 * 32];   // emb_v transposed: [d][dist]
  __shared__ _Float16 qs [4 * 16 * 64]; // per-wave Q tile
  __shared__ _Float16 ks_[4 * 16 * 64]; // per-wave K tile
  __shared__ _Float16 vts[4 * 64 * 16]; // per-wave V transposed: [d][s]
  __shared__ float    scr[4 * 16 * 32]; // per-wave scratch: P then attn

  const int tid  = threadIdx.x;
  const int lane = tid & 31;
  const int w    = tid >> 5;
  const int bh   = blockIdx.x * 4 + w;  // 32768 (b,h) pairs total
  const int mrow = lane & 15;
  const int lh   = lane >> 4;

  // shared emb staging (block-wide)
  for (int i = tid; i < 2048; i += 128) {
    ek_s[i] = ekh[i];                       // dist 0..31 only (32 never occurs)
    int dist = i >> 6, d = i & 63;
    evt_s[d * 32 + dist] = evh[i];
  }
  // per-wave Q/K staging: async global->LDS b128 copies
  {
    const uint4* qsrc = (const uint4*)(qg + (size_t)bh * 1024);
    const uint4* ksrc = (const uint4*)(kg + (size_t)bh * 1024);
    uint4* qdst = (uint4*)&qs [w * 1024];
    uint4* kdst = (uint4*)&ks_[w * 1024];
    for (int i = lane; i < 128; i += 32) {
      async_copy_b128(qsrc + i, qdst + i);
      async_copy_b128(ksrc + i, kdst + i);
    }
    // V needs a transpose -> scalar DS stores
    const _Float16* vsrc = vg + (size_t)bh * 1024;
    _Float16* vdst = &vts[w * 1024];
    for (int i = lane; i < 1024; i += 32) {
      int s = i >> 6, d = i & 63;
      vdst[d * 16 + s] = vsrc[i];           // transpose to [d][s]
    }
  }
  async_wait0();
  __syncthreads();

  const _Float16* qrow = &qs[w * 1024 + mrow * 64];
  v16h qa0 = ld_frag(qrow, 0, lh);
  v16h qa1 = ld_frag(qrow, 32, lh);

  // sim = Q K^T (16x16, K-dim 64 -> 2 WMMA)
  v8f sim = {};
  {
    const _Float16* krow = &ks_[w * 1024 + mrow * 64];
    sim = wmma16(qa0, ld_frag(krow, 0, lh), sim);
    sim = wmma16(qa1, ld_frag(krow, 32, lh), sim);
  }
  // P = Q @ emb_k[0:32]^T  (16x32 -> 4 WMMA)
  v8f p0 = {}, p1 = {};
  {
    const _Float16* e0 = &ek_s[mrow * 64];
    const _Float16* e1 = &ek_s[(16 + mrow) * 64];
    p0 = wmma16(qa0, ld_frag(e0, 0, lh), p0);
    p0 = wmma16(qa1, ld_frag(e0, 32, lh), p0);
    p1 = wmma16(qa0, ld_frag(e1, 0, lh), p1);
    p1 = wmma16(qa1, ld_frag(e1, 32, lh), p1);
  }

  // stash P[m][dist] in wave-private scratch (stride 32)
  float* sw = &scr[w * 512];
#pragma unroll
  for (int r = 0; r < 8; ++r) {
    int m = r + 8 * lh;
    sw[m * 32 + mrow]      = p0[r];
    sw[m * 32 + 16 + mrow] = p1[r];
  }
  asm volatile("s_wait_dscnt 0" ::: "memory");   // wave-private LDS RAW fence

  // bias gather (dist = s-m+16), scale, row softmax over 16-lane groups
  float attnv[8];
#pragma unroll
  for (int r = 0; r < 8; ++r) {
    int m = r + 8 * lh;
    float bias = sw[m * 32 + (mrow - m + 16)];   // in [1,31] always
    float sv = (sim[r] + bias) * ATT_SCALE;
    float mx = sv;
#pragma unroll
    for (int off = 8; off >= 1; off >>= 1) mx = fmaxf(mx, __shfl_xor(mx, off, 16));
    float e = __expf(sv - mx);
    float sum = e;
#pragma unroll
    for (int off = 8; off >= 1; off >>= 1) sum += __shfl_xor(sum, off, 16);
    attnv[r] = e / sum;
  }
  asm volatile("s_wait_dscnt 0" ::: "memory");   // WAR: bias reads before reuse

  // write attn[m][s] (stride 16) into same scratch
#pragma unroll
  for (int r = 0; r < 8; ++r) sw[(r + 8 * lh) * 16 + mrow] = attnv[r];
  asm volatile("s_wait_dscnt 0" ::: "memory");

  // A-fragment: attn (16x32, K = s, upper 16 of K zero-padded)
  v16h aa;
  {
    union { v16h v; _Float16 h[16]; } u;
    const float* arow = &sw[mrow * 16 + 8 * lh];
    float4 f0 = *(const float4*)arow;
    float4 f1 = *(const float4*)(arow + 4);
    u.h[0] = (_Float16)f0.x; u.h[1] = (_Float16)f0.y;
    u.h[2] = (_Float16)f0.z; u.h[3] = (_Float16)f0.w;
    u.h[4] = (_Float16)f1.x; u.h[5] = (_Float16)f1.y;
    u.h[6] = (_Float16)f1.z; u.h[7] = (_Float16)f1.w;
#pragma unroll
    for (int e = 8; e < 16; ++e) u.h[e] = (_Float16)0.0f;
    aa = u.v;
  }
  // A-fragment: Arow (16x32, K = dist j; Arow[m][j] = attn[m][j+m-16] or 0)
  v16h ar;
  {
    union { v16h v; _Float16 h[16]; } u;
#pragma unroll
    for (int e = 0; e < 16; ++e) {
      int j = (e < 8) ? (8 * lh + e) : (8 * lh + e + 8);
      int s = j + mrow - 16;
      u.h[e] = (s >= 0 && s < 16) ? (_Float16)sw[mrow * 16 + s] : (_Float16)0.0f;
    }
    ar = u.v;
  }

  // out = attn @ V + Arow @ emb_v[0:32]   (4 d-tiles x 2 WMMA)
  const int b = bh >> 3, hi = bh & 7;
#pragma unroll
  for (int nt = 0; nt < 4; ++nt) {
    int dcol = nt * 16 + mrow;
    v8f o = {};
    v16h vbfr;
    {
      union { v16h v; uint4 q[2]; } u;
      u.q[0] = *(const uint4*)&vts[w * 1024 + dcol * 16 + 8 * lh]; // s = 8lh..
      u.q[1] = make_uint4(0u, 0u, 0u, 0u);                         // K>=16: A is 0
      vbfr = u.v;
    }
    o = wmma16(aa, vbfr, o);
    o = wmma16(ar, ld_frag(&evt_s[dcol * 32], 0, lh), o);
#pragma unroll
    for (int r = 0; r < 8; ++r) {
      int m = r + 8 * lh;
      outh[(size_t)(b * 16 + m) * 512 + hi * 64 + dcol] = (_Float16)o[r];
    }
  }
}

// --- host launcher ----------------------------------------------------------

static inline size_t align256(size_t x) { return (x + 255) & ~(size_t)255; }

extern "C" void kernel_launch(void* const* d_in, const int* in_sizes, int n_in,
                              void* d_out, int out_size, void* d_ws, size_t ws_size,
                              hipStream_t stream) {
  const float* x     = (const float*)d_in[0];
  const float* Wq    = (const float*)d_in[1];
  const float* Wk    = (const float*)d_in[2];
  const float* Wv    = (const float*)d_in[3];
  const float* emb_k = (const float*)d_in[4];
  const float* emb_v = (const float*)d_in[5];
  const float* Wo    = (const float*)d_in[6];
  const float* bo    = (const float*)d_in[7];
  float* out = (float*)d_out;

  // workspace carve (all f16 intermediates; ~270 MB total)
  char* ws = (char*)d_ws;
  size_t off = 0;
  _Float16* wqkv_t = (_Float16*)(ws + off); off = align256(off + (size_t)1536 * 512 * 2);
  _Float16* wo_t   = (_Float16*)(ws + off); off = align256(off + (size_t)512 * 512 * 2);
  _Float16* ekh    = (_Float16*)(ws + off); off = align256(off + (size_t)33 * 64 * 2);
  _Float16* evh    = (_Float16*)(ws + off); off = align256(off + (size_t)33 * 64 * 2);
  const size_t qkv_elems = (size_t)B_TOT * N_HEADS * N_TOK * D_HEAD;   // 33.5M
  _Float16* qb     = (_Float16*)(ws + off); off = align256(off + qkv_elems * 2);
  _Float16* kb     = (_Float16*)(ws + off); off = align256(off + qkv_elems * 2);
  _Float16* vb     = (_Float16*)(ws + off); off = align256(off + qkv_elems * 2);
  _Float16* attn_h = (_Float16*)(ws + off); off = align256(off + (size_t)M_TOK * D_MODEL * 2);
  (void)ws_size; (void)n_in; (void)in_sizes; (void)out_size;

  // 1) pack weights / embeddings to f16 (N-major for GEMM B)
  pack_wqkv_kernel<<<(1536 * 512 + 255) / 256, 256, 0, stream>>>(Wq, Wk, Wv, wqkv_t);
  pack_wo_kernel<<<(512 * 512 + 255) / 256, 256, 0, stream>>>(Wo, wo_t);
  pack_emb_kernel<<<(33 * 64 + 255) / 256, 256, 0, stream>>>(emb_k, emb_v, ekh, evh);

  // 2) fused QKV projection: (65536 x 512) * (512 x 1536) -> head-split f16
  gemm_wmma_kernel<0><<<dim3(1536 / 64, M_TOK / 128), 256, 0, stream>>>(
      (const void*)x, wqkv_t, nullptr, qb, kb, vb, nullptr, M_TOK, 1536, 512);

  // 3) attention core: one wave per (b,h); 4 waves per block
  attention_kernel<<<(B_TOT * N_HEADS) / 4, 128, 0, stream>>>(
      qb, kb, vb, ekh, evh, attn_h);

  // 4) output projection + bias: (65536 x 512) * (512 x 512) -> f32 out
  gemm_wmma_kernel<1><<<dim3(512 / 64, M_TOK / 128), 256, 0, stream>>>(
      (const void*)attn_h, wo_t, bo, nullptr, nullptr, nullptr, out, M_TOK, 512, 512);
}
